// SwinBlock_55327768708009
// MI455X (gfx1250) — matrix operand order repaired and checked by reading, not compiled, not run on hardware
//
#include <hip/hip_runtime.h>
#include <hip/hip_bf16.h>

typedef __attribute__((ext_vector_type(8)))  _Float16 v8h;
typedef __attribute__((ext_vector_type(16))) _Float16 v16h;
typedef __attribute__((ext_vector_type(8)))  float    v8f;

#define TOKENS   100352      // 32 * 3136 == 2048 windows * 49
#define DIM      512
#define HEADS    16
#define HD       32
#define NWIN     2048        // 32 batches * 64 windows
#define QKVW     1536
#define HIDDEN   2048

// map a window-order row (w = window id, t = token in window) to the original
// image token index (undo window partition + cyclic shift by +3)
__device__ __forceinline__ size_t win_row_to_img(int row) {
    int w = row / 49, t = row - w * 49;
    int b = w >> 6, iw = w & 63;
    int th = t / 7, tw = t - th * 7;
    int sh = (iw >> 3) * 7 + th;
    int sw = (iw & 7) * 7 + tw;
    int gh = sh + 3; if (gh >= 56) gh -= 56;
    int gw = sw + 3; if (gw >= 56) gw -= 56;
    return ((size_t)b * 3136 + (size_t)gh * 56 + gw);
}

// -------- weight conversion f32 -> f16 --------
__global__ void f2h_kernel(const float* __restrict__ s, _Float16* __restrict__ d, int n) {
    int i = blockIdx.x * 256 + threadIdx.x;
    if (i < n) d[i] = (_Float16)s[i];
}

// -------- precompute attention bias plane (per head) and mask planes (4 edge types) ----
// PB[h][m][n]: rel-pos bias for m,n<49 ; -1e30 for n>=49 (key padding) ; 0 for m>=49
// MB[t][m][n]: shift-window mask (0 / -100), t = edgeH*2+edgeW
__global__ void build_tables(const float* __restrict__ rel_bias,
                             float* __restrict__ PB, float* __restrict__ MB) {
    int i = blockIdx.x * 256 + threadIdx.x;
    if (i < HEADS * 64 * 64) {
        int h = i >> 12, m = (i >> 6) & 63, n = i & 63;
        float v;
        if (n >= 49)      v = -1e30f;
        else if (m >= 49) v = 0.f;
        else {
            int mh = m / 7, mw = m - mh * 7, nh = n / 7, nw = n - nh * 7;
            v = rel_bias[((mh - nh + 6) * 13 + (mw - nw + 6)) * HEADS + h];
        }
        PB[i] = v;
    }
    if (i < 4 * 64 * 64) {
        int t = i >> 12, m = (i >> 6) & 63, n = i & 63;
        float v = 0.f;
        if (m < 49 && n < 49) {
            int whh = (t & 2) ? 7 : 0, www = (t & 1) ? 7 : 0;
            int gmh = whh * 7 + m / 7, gmw = www * 7 + m % 7;
            int gnh = whh * 7 + n / 7, gnw = www * 7 + n % 7;
            int lm  = (gmh < 49 ? 0 : (gmh < 53 ? 1 : 2)) * 3 + (gmw < 49 ? 0 : (gmw < 53 ? 1 : 2));
            int ln_ = (gnh < 49 ? 0 : (gnh < 53 ? 1 : 2)) * 3 + (gnw < 49 ? 0 : (gnw < 53 ? 1 : 2));
            if (lm != ln_) v = -100.f;
        }
        MB[i] = v;
    }
}

// -------- LayerNorm (optionally gathering through shift+window map) --------
__global__ __launch_bounds__(256)
void ln_kernel(const float* __restrict__ X, const float* __restrict__ g,
               const float* __restrict__ be, _Float16* __restrict__ out, int gather) {
    int row  = blockIdx.x * 8 + (threadIdx.x >> 5);
    int lane = threadIdx.x & 31;
    if (row >= TOKENS) return;
    size_t src = gather ? (win_row_to_img(row) * DIM) : ((size_t)row * DIM);
    const float* xp = X + src + lane * 16;
    float vals[16];
    float s = 0.f, sq = 0.f;
#pragma unroll
    for (int i = 0; i < 16; ++i) { float v = xp[i]; vals[i] = v; s += v; sq += v * v; }
#pragma unroll
    for (int o = 16; o > 0; o >>= 1) { s += __shfl_xor(s, o, 32); sq += __shfl_xor(sq, o, 32); }
    float mean = s * (1.f / DIM);
    float var  = sq * (1.f / DIM) - mean * mean;
    float rstd = rsqrtf(var + 1e-5f);
    _Float16* op = out + (size_t)row * DIM + lane * 16;
#pragma unroll
    for (int i = 0; i < 16; ++i) {
        int c = lane * 16 + i;
        op[i] = (_Float16)((vals[i] - mean) * rstd * g[c] + be[c]);
    }
}

// -------- generic f16 WMMA GEMM: out = A[M,K] * W[N,K]^T, per-wave 32x64 tile --------
// EPI 0: qkv  (add bias, store f16)
// EPI 1: proj (add bias + shortcut, scatter window-reverse+roll, store f32)
// EPI 2: fc1  (add bias, exact GELU, store f16)
// EPI 3: fc2  (add bias, accumulate into f32 out)
template <int EPI>
__global__ __launch_bounds__(256)
void gemm_epi(const _Float16* __restrict__ A, const _Float16* __restrict__ Bw,
              const float* __restrict__ bias, int M, int N, int K,
              _Float16* __restrict__ outH, float* __restrict__ outF,
              const float* __restrict__ addF) {
    int wid  = blockIdx.x * 8 + (threadIdx.x >> 5);
    int lane = threadIdx.x & 31;
    int ngrp = N >> 6;
    int total = (M >> 5) * ngrp;
    if (wid >= total) return;
    int mt = wid / ngrp, ng = wid - mt * ngrp;
    int mbase = mt << 5, nbase = ng << 6;

    int koff = (lane & 16) ? 8 : 0;     // A K sub-chunk (ISA 16-bit A layout)
    int boff = (lane & 16) ? 8 : 0;     // B K sub-chunk (same order as A)
    const _Float16* arp0 = A + (size_t)(mbase + (lane & 15)) * K;
    const _Float16* arp1 = arp0 + (size_t)16 * K;

    v8f acc[2][4] = {};
    for (int k0 = 0; k0 < K; k0 += 32) {
        v16h a0, a1;
        *(v8h*)&a0       = *(const v8h*)(arp0 + k0 + koff);
        *((v8h*)&a0 + 1) = *(const v8h*)(arp0 + k0 + koff + 16);
        *(v8h*)&a1       = *(const v8h*)(arp1 + k0 + koff);
        *((v8h*)&a1 + 1) = *(const v8h*)(arp1 + k0 + koff + 16);
#pragma unroll
        for (int j = 0; j < 4; ++j) {
            int bn = nbase + (j << 4) + (lane & 15);
            const _Float16* brp = Bw + (size_t)bn * K + k0 + boff;
            v16h b;
            *(v8h*)&b       = *(const v8h*)(brp);
            *((v8h*)&b + 1) = *(const v8h*)(brp + 16);
            acc[0][j] = __builtin_amdgcn_wmma_f32_16x16x32_f16(
                false, a0, false, b, (short)0, acc[0][j], false, false);
            acc[1][j] = __builtin_amdgcn_wmma_f32_16x16x32_f16(
                false, a1, false, b, (short)0, acc[1][j], false, false);
        }
    }

    float bv[4];
#pragma unroll
    for (int j = 0; j < 4; ++j) bv[j] = bias[nbase + (j << 4) + (lane & 15)];

#pragma unroll
    for (int i = 0; i < 2; ++i) {
#pragma unroll
        for (int r = 0; r < 8; ++r) {
            int m = mbase + (i << 4) + r + ((lane & 16) ? 8 : 0);
            size_t rowbase;
            if (EPI == 1)      rowbase = win_row_to_img(m) * DIM;
            else               rowbase = (size_t)m * N;
#pragma unroll
            for (int j = 0; j < 4; ++j) {
                int n = nbase + (j << 4) + (lane & 15);
                float v = acc[i][j][r] + bv[j];
                if (EPI == 0) {
                    outH[rowbase + n] = (_Float16)v;
                } else if (EPI == 1) {
                    size_t idx = rowbase + n;
                    outF[idx] = v + addF[idx];
                } else if (EPI == 2) {
                    float ge = 0.5f * v * (1.0f + erff(v * 0.70710678118654752f));
                    outH[rowbase + n] = (_Float16)ge;
                } else {
                    size_t idx = rowbase + n;   // N == DIM here
                    outF[idx] = outF[idx] + v;
                }
            }
        }
    }
}

// -------- windowed attention: one block = one (window, head) --------
__global__ __launch_bounds__(128)
void attn_kernel(const _Float16* __restrict__ qkv, const float* __restrict__ PB,
                 const float* __restrict__ MB, _Float16* __restrict__ ob) {
    __shared__ float    S[64][65];
    __shared__ _Float16 P[64][72];
    __shared__ _Float16 Vs[64][36];
    int bid  = blockIdx.x;
    int win  = bid >> 4, h = bid & 15;
    int iw   = win & 63;
    int lane = threadIdx.x & 31, wv = threadIdx.x >> 5;
    size_t base = (size_t)win * 49 * QKVW + h * HD;
    const _Float16* qp = qkv + base;
    const _Float16* kp = qkv + base + DIM;
    const _Float16* vp = qkv + base + 2 * DIM;
    int koff = (lane & 16) ? 8 : 0;

    // stage V into LDS (rows >=49 zero-filled), coalesced
    for (int idx = threadIdx.x; idx < 64 * 32; idx += 128) {
        int tok = idx >> 5, d = idx & 31;
        Vs[tok][d] = (tok < 49) ? vp[(size_t)tok * QKVW + d] : (_Float16)0.f;
    }

    // ---- S = scale * q @ k^T + bias + mask (tables) ----
    int am = (wv << 4) + (lane & 15);
    v16h a = {};
    if (am < 49) {
        const _Float16* ap = qp + (size_t)am * QKVW + koff;
        *(v8h*)&a       = *(const v8h*)(ap);
        *((v8h*)&a + 1) = *(const v8h*)(ap + 16);
    }
    int type = (((iw >> 3) == 7) ? 2 : 0) | (((iw & 7) == 7) ? 1 : 0);
    const float* pb = PB + ((size_t)h << 12);
    const float* mb = MB + ((size_t)type << 12);
#pragma unroll
    for (int j = 0; j < 4; ++j) {
        int bn = (j << 4) + (lane & 15);
        v16h b = {};
        if (bn < 49) {
            const _Float16* bp = kp + (size_t)bn * QKVW + ((lane & 16) ? 8 : 0);
            *(v8h*)&b       = *(const v8h*)(bp);
            *((v8h*)&b + 1) = *(const v8h*)(bp + 16);
        }
        v8f acc = {};
        acc = __builtin_amdgcn_wmma_f32_16x16x32_f16(
            false, a, false, b, (short)0, acc, false, false);
#pragma unroll
        for (int r = 0; r < 8; ++r) {
            int m = (wv << 4) + r + ((lane & 16) ? 8 : 0);
            int n = (j << 4) + (lane & 15);
            int t = (m << 6) + n;
            S[m][n] = fmaf(acc[r], 0.17677669529663687f, pb[t] + mb[t]);
        }
    }
    __syncthreads();

    // ---- softmax rows ----
    if (threadIdx.x < 64) {
        int m = threadIdx.x;
        float mx = -1e30f;
        for (int n = 0; n < 64; ++n) mx = fmaxf(mx, S[m][n]);
        float sum = 0.f;
        for (int n = 0; n < 64; ++n) sum += __expf(S[m][n] - mx);
        float inv = 1.0f / sum;
        for (int n = 0; n < 64; ++n) P[m][n] = (_Float16)(__expf(S[m][n] - mx) * inv);
    }
    __syncthreads();

    // ---- O = P @ V ----
#pragma unroll
    for (int j = 0; j < 2; ++j) {
        v8f acc = {};
#pragma unroll
        for (int ks = 0; ks < 2; ++ks) {
            int k0 = ks << 5;
            int mrow = (wv << 4) + (lane & 15);
            v16h a2;
#pragma unroll
            for (int i = 0; i < 8; ++i) {
                a2[i]     = P[mrow][k0 + koff + i];
                a2[i + 8] = P[mrow][k0 + koff + 16 + i];
            }
            int n  = (j << 4) + (lane & 15);
            int kb = k0 + ((lane & 16) ? 8 : 0);
            v16h b2;
#pragma unroll
            for (int i = 0; i < 16; ++i) b2[i] = Vs[kb + i + ((i >= 8) ? 8 : 0)][n];
            acc = __builtin_amdgcn_wmma_f32_16x16x32_f16(
                false, a2, false, b2, (short)0, acc, false, false);
        }
#pragma unroll
        for (int r = 0; r < 8; ++r) {
            int m = (wv << 4) + r + ((lane & 16) ? 8 : 0);
            if (m < 49) {
                int d = (j << 4) + (lane & 15);
                ob[((size_t)win * 49 + m) * DIM + h * HD + d] = (_Float16)acc[r];
            }
        }
    }
}

extern "C" void kernel_launch(void* const* d_in, const int* in_sizes, int n_in,
                              void* d_out, int out_size, void* d_ws, size_t ws_size,
                              hipStream_t stream) {
    (void)in_sizes; (void)n_in; (void)out_size; (void)ws_size;
    const float* x      = (const float*)d_in[0];
    const float* n1g    = (const float*)d_in[1];
    const float* n1b    = (const float*)d_in[2];
    const float* qkv_w  = (const float*)d_in[3];
    const float* qkv_b  = (const float*)d_in[4];
    const float* relb   = (const float*)d_in[5];
    const float* proj_w = (const float*)d_in[6];
    const float* proj_b = (const float*)d_in[7];
    const float* n2g    = (const float*)d_in[8];
    const float* n2b    = (const float*)d_in[9];
    const float* fc1_w  = (const float*)d_in[10];
    const float* fc1_b  = (const float*)d_in[11];
    const float* fc2_w  = (const float*)d_in[12];
    const float* fc2_b  = (const float*)d_in[13];
    float* out = (float*)d_out;

    char* ws = (char*)d_ws;
    size_t off = 0;
    auto alloc = [&](size_t bytes) -> void* {
        void* p = ws + off;
        off += (bytes + 255) & ~(size_t)255;
        return p;
    };
    _Float16* Wq   = (_Float16*)alloc((size_t)QKVW * DIM * 2);
    _Float16* Wp   = (_Float16*)alloc((size_t)DIM * DIM * 2);
    _Float16* W1   = (_Float16*)alloc((size_t)HIDDEN * DIM * 2);
    _Float16* W2   = (_Float16*)alloc((size_t)DIM * HIDDEN * 2);
    float*    PB   = (float*)alloc((size_t)HEADS * 64 * 64 * 4);
    float*    MB   = (float*)alloc((size_t)4 * 64 * 64 * 4);
    _Float16* yw   = (_Float16*)alloc((size_t)TOKENS * DIM * 2);    // LN1 out / LN2 out
    _Float16* qkvb = (_Float16*)alloc((size_t)TOKENS * QKVW * 2);
    _Float16* ob   = (_Float16*)alloc((size_t)TOKENS * DIM * 2);
    _Float16* hmid = (_Float16*)alloc((size_t)TOKENS * HIDDEN * 2);

    // 1) weight conversion + attention tables
    f2h_kernel<<<(QKVW * DIM + 255) / 256, 256, 0, stream>>>(qkv_w, Wq, QKVW * DIM);
    f2h_kernel<<<(DIM * DIM + 255) / 256, 256, 0, stream>>>(proj_w, Wp, DIM * DIM);
    f2h_kernel<<<(HIDDEN * DIM + 255) / 256, 256, 0, stream>>>(fc1_w, W1, HIDDEN * DIM);
    f2h_kernel<<<(DIM * HIDDEN + 255) / 256, 256, 0, stream>>>(fc2_w, W2, DIM * HIDDEN);
    build_tables<<<(HEADS * 64 * 64 + 255) / 256, 256, 0, stream>>>(relb, PB, MB);

    // 2) LN1 + shift + window partition (gather)
    ln_kernel<<<(TOKENS + 7) / 8, 256, 0, stream>>>(x, n1g, n1b, yw, 1);

    // 3) QKV GEMM: M=TOKENS, N=1536, K=512
    {
        int waves = (TOKENS / 32) * (QKVW / 64);
        gemm_epi<0><<<(waves + 7) / 8, 256, 0, stream>>>(
            yw, Wq, qkv_b, TOKENS, QKVW, DIM, qkvb, nullptr, nullptr);
    }

    // 4) windowed attention
    attn_kernel<<<NWIN * HEADS, 128, 0, stream>>>(qkvb, PB, MB, ob);

    // 5) proj GEMM + window reverse + roll + residual -> d_out = x2
    {
        int waves = (TOKENS / 32) * (DIM / 64);
        gemm_epi<1><<<(waves + 7) / 8, 256, 0, stream>>>(
            ob, Wp, proj_b, TOKENS, DIM, DIM, nullptr, out, x);
    }

    // 6) LN2 (identity mapping)
    ln_kernel<<<(TOKENS + 7) / 8, 256, 0, stream>>>(out, n2g, n2b, yw, 0);

    // 7) FC1 GEMM + GELU: N=2048, K=512
    {
        int waves = (TOKENS / 32) * (HIDDEN / 64);
        gemm_epi<2><<<(waves + 7) / 8, 256, 0, stream>>>(
            yw, W1, fc1_b, TOKENS, HIDDEN, DIM, hmid, nullptr, nullptr);
    }

    // 8) FC2 GEMM + residual accumulate: N=512, K=2048
    {
        int waves = (TOKENS / 32) * (DIM / 64);
        gemm_epi<3><<<(waves + 7) / 8, 256, 0, stream>>>(
            hmid, W2, fc2_b, TOKENS, DIM, HIDDEN, nullptr, out, nullptr);
    }
}
